// FlaxFalconAttention_69303592288754
// MI455X (gfx1250) — matrix operand order repaired and checked by reading, not compiled, not run on hardware
//
#include <hip/hip_runtime.h>
#include <hip/hip_bf16.h>

typedef __attribute__((ext_vector_type(16))) __bf16 v16bf;
typedef __attribute__((ext_vector_type(8)))  float  v8f;
typedef __attribute__((ext_vector_type(8)))  unsigned short ushort8;
typedef __attribute__((ext_vector_type(16))) unsigned short ushort16;
typedef __attribute__((ext_vector_type(4)))  int v4i;

#define HID   4544
#define NHEADS  71
#define HD      64
#define SEQ   1024
#define QKVN  4672   // HID + 2*HD

// CDNA5 async global->LDS DMA (ASYNCcnt path), if the toolchain exposes it.
#if __has_builtin(__builtin_amdgcn_global_load_async_to_lds_b128) && \
    __has_builtin(__builtin_amdgcn_s_wait_asynccnt)
#define USE_ASYNC_LDS 1
#else
#define USE_ASYNC_LDS 0
#endif

// ---------- helpers ----------

__device__ __forceinline__ unsigned short f2bf(float f) {
  unsigned int u = __float_as_uint(f);
  u += 0x7FFFu + ((u >> 16) & 1u);      // round-to-nearest-even
  return (unsigned short)(u >> 16);
}

// Build a 16-element bf16 fragment from two 16-byte chunks.
__device__ __forceinline__ v16bf frag16(const unsigned short* p0,
                                        const unsigned short* p1) {
  ushort8 a = *(const ushort8*)p0;
  ushort8 b = *(const ushort8*)p1;
  ushort16 u;
#pragma unroll
  for (int i = 0; i < 8; ++i) { u[i] = a[i]; u[i + 8] = b[i]; }
  return __builtin_bit_cast(v16bf, u);
}

__device__ __forceinline__ v8f wmma_bf16(v16bf a, v16bf b, v8f c) {
  return __builtin_amdgcn_wmma_f32_16x16x32_bf16(false, a, false, b,
                                                 (short)0, c, false, false);
}

#if USE_ASYNC_LDS
__device__ __forceinline__ void async_b128(const unsigned short* g,
                                           unsigned short* l) {
  __builtin_amdgcn_global_load_async_to_lds_b128(
      (__attribute__((address_space(1))) v4i*)(unsigned short*)g,
      (__attribute__((address_space(3))) v4i*)l, 0, 0);
}
#endif

// ---------- conversion kernels ----------

__global__ void __launch_bounds__(256) cvt_bf16(const float* __restrict__ in,
                                                unsigned short* __restrict__ out,
                                                int n) {
  int i = blockIdx.x * blockDim.x + threadIdx.x;
  if (i < n) out[i] = f2bf(in[i]);
}

// W[K][N] fp32 -> Wt[N][K] bf16
__global__ void __launch_bounds__(256) cvt_transpose(const float* __restrict__ W,
                                                     unsigned short* __restrict__ Wt,
                                                     int K, int N) {
  long long total = (long long)K * N;
  for (long long idx = (long long)blockIdx.x * blockDim.x + threadIdx.x;
       idx < total; idx += (long long)gridDim.x * blockDim.x) {
    long long n = idx / K, k = idx - n * K;
    Wt[idx] = f2bf(W[k * (long long)N + n]);
  }
}

// ---------- GEMM: C[M][N] = A[M][K] * Bt[N][K]^T  (bf16 in, f32 out) ----------
// Block tile 128x64, BK=32, 256 threads = 8 waves (4 in M x 2 in N);
// each wave computes a 32x32 macro-tile = 4 WMMAs per K-step.

__global__ void __launch_bounds__(256) gemm_bf16_tn(
    const unsigned short* __restrict__ A,
    const unsigned short* __restrict__ Bt,
    float* __restrict__ C, int M, int N, int K) {
  __shared__ __align__(16) unsigned short sA[128][32];
  __shared__ __align__(16) unsigned short sB[64][32];

  const int tid  = threadIdx.x;
  const int lane = tid & 31;
  const int w    = tid >> 5;
  const int lo   = lane & 15, hi = lane >> 4;
  const int wm   = w & 3,     wn = w >> 2;
  const int bm   = blockIdx.y * 128, bn = blockIdx.x * 64;

  // cooperative loads: thread -> 16B chunks (2 for A, 1 for B)
  const int lr = tid >> 2, lc = (tid & 3) * 8;
  const unsigned short* aG0 = A  + (size_t)(bm + lr) * K + lc;
  const unsigned short* aG1 = A  + (size_t)(bm + 64 + lr) * K + lc;
  const unsigned short* bG  = Bt + (size_t)(bn + lr) * K + lc;

  const v8f vzero = {0.f,0.f,0.f,0.f,0.f,0.f,0.f,0.f};
  v8f acc00 = vzero, acc01 = vzero, acc10 = vzero, acc11 = vzero;

  for (int kk = 0; kk < K; kk += 32) {
    __syncthreads();
#if USE_ASYNC_LDS
    async_b128(aG0 + kk, &sA[lr][lc]);
    async_b128(aG1 + kk, &sA[64 + lr][lc]);
    async_b128(bG + kk, &sB[lr][lc]);
#else
    *(ushort8*)&sA[lr][lc]      = *(const ushort8*)(aG0 + kk);
    *(ushort8*)&sA[64 + lr][lc] = *(const ushort8*)(aG1 + kk);
    *(ushort8*)&sB[lr][lc]      = *(const ushort8*)(bG + kk);
#endif
    if (kk + 32 < K) {                       // global_prefetch_b8 of next tiles
      __builtin_prefetch(aG0 + kk + 32, 0, 1);
      __builtin_prefetch(aG1 + kk + 32, 0, 1);
      __builtin_prefetch(bG + kk + 32, 0, 1);
    }
#if USE_ASYNC_LDS
    __builtin_amdgcn_s_wait_asynccnt(0);
#endif
    __syncthreads();

    // A frags: m = wm*32 + i*16 + lo, k = 8*hi+e (e<8) / 16+8*hi+(e-8)
    v16bf af0 = frag16(&sA[wm * 32 + lo][8 * hi],
                       &sA[wm * 32 + lo][16 + 8 * hi]);
    v16bf af1 = frag16(&sA[wm * 32 + 16 + lo][8 * hi],
                       &sA[wm * 32 + 16 + lo][16 + 8 * hi]);
    // B frags: n = wn*32 + j*16 + lo, k = 8*hi+e (e<8) / 16+8*hi+(e-8)
    v16bf bf0 = frag16(&sB[wn * 32 + lo][8 * hi],
                       &sB[wn * 32 + lo][16 + 8 * hi]);
    v16bf bf1 = frag16(&sB[wn * 32 + 16 + lo][8 * hi],
                       &sB[wn * 32 + 16 + lo][16 + 8 * hi]);
    acc00 = wmma_bf16(af0, bf0, acc00);
    acc01 = wmma_bf16(af0, bf1, acc01);
    acc10 = wmma_bf16(af1, bf0, acc10);
    acc11 = wmma_bf16(af1, bf1, acc11);
  }

#pragma unroll
  for (int r = 0; r < 8; ++r) {
    int row0 = bm + wm * 32 + r + 8 * hi;     // C layout: m = r + 8*hi
    float* c0 = C + (size_t)row0 * N + bn + wn * 32 + lo;
    float* c1 = C + (size_t)(row0 + 16) * N + bn + wn * 32 + lo;
    c0[0]  = acc00[r];
    c0[16] = acc01[r];
    c1[0]  = acc10[r];
    c1[16] = acc11[r];
  }
}

// ---------- RoPE + split fused -> q(bf16), k(bf16), v^T(bf16) ----------

__global__ void __launch_bounds__(256) rope_split(
    const float* __restrict__ fused, const int* __restrict__ pos_ids,
    unsigned short* __restrict__ qb,   // [SEQ][HID]
    unsigned short* __restrict__ kb,   // [SEQ][HD]
    unsigned short* __restrict__ vT) { // [HD][SEQ]
  const int s = blockIdx.x;
  const float pos = (float)pos_ids[s];
  const float* row = fused + (size_t)s * QKVN;
  const float LOG1E4_OVER_32 = 0.28782313662425572f;  // ln(10000)/32

  for (int idx = threadIdx.x; idx < NHEADS * 32; idx += blockDim.x) {
    int hh = idx >> 5, d = idx & 31;
    float ang = pos * __expf(-(float)d * LOG1E4_OVER_32);
    float sn, cs; __sincosf(ang, &sn, &cs);
    float x0 = row[hh * HD + d], x1 = row[hh * HD + d + 32];
    qb[(size_t)s * HID + hh * HD + d]      = f2bf(x0 * cs - x1 * sn);
    qb[(size_t)s * HID + hh * HD + d + 32] = f2bf(x1 * cs + x0 * sn);
  }
  if (threadIdx.x < 32) {
    int d = threadIdx.x;
    float ang = pos * __expf(-(float)d * LOG1E4_OVER_32);
    float sn, cs; __sincosf(ang, &sn, &cs);
    float x0 = row[HID + d], x1 = row[HID + d + 32];
    kb[s * HD + d]      = f2bf(x0 * cs - x1 * sn);
    kb[s * HD + d + 32] = f2bf(x1 * cs + x0 * sn);
  }
  if (threadIdx.x < HD) {
    vT[(size_t)threadIdx.x * SEQ + s] = f2bf(row[HID + HD + threadIdx.x]);
  }
}

// ---------- flash attention (multi-query, causal) ----------
// grid = (SEQ/128, NHEADS); 8 waves/block; each wave owns 16 queries of one head.
// Computes S^T = K*Q^T so P lands directly in A-fragment layout for P*V.

__global__ void __launch_bounds__(256) falcon_attn(
    const unsigned short* __restrict__ qb,  // [SEQ][HID]
    const unsigned short* __restrict__ kb,  // [SEQ][HD]
    const unsigned short* __restrict__ vT,  // [HD][SEQ]
    unsigned short* __restrict__ ctxb) {    // [SEQ][HID]
  const int lane = threadIdx.x & 31, w = threadIdx.x >> 5;
  const int lo = lane & 15, hi = lane >> 4;
  const int h = blockIdx.y;
  const int qbase = blockIdx.x * 128 + w * 16;
  const int q = qbase + lo;                 // this lane's query (as S^T column)

  // Q^T B-fragments (held for whole key loop): n=query=lo, k=dim=8*hi+e / 16+8*hi+e
  const unsigned short* qrow = qb + (size_t)q * HID + h * HD;
  v16bf qf0 = frag16(qrow + 8 * hi,      qrow + 16 + 8 * hi);
  v16bf qf1 = frag16(qrow + 32 + 8 * hi, qrow + 48 + 8 * hi);

  const v8f vzero = {0.f,0.f,0.f,0.f,0.f,0.f,0.f,0.f};
  v8f o[4] = {vzero, vzero, vzero, vzero}; // 16q x 64d accumulator, 4 N-chunks
  float mrun = -1e30f, srun = 0.f;
  const float SC = 0.125f;                  // 1/sqrt(64)

  for (int kt = 0; kt < qbase + 16; kt += 32) {
    // K A-fragments: m = key = lo (per 16-key tile), k = dim = 8*hi+e / 16+8*hi+e
    const unsigned short* k0 = kb + (size_t)(kt + lo) * HD;
    const unsigned short* k1 = kb + (size_t)(kt + 16 + lo) * HD;
    v8f st0 = wmma_bf16(frag16(k0 + 8 * hi, k0 + 16 + 8 * hi), qf0, vzero);
    st0     = wmma_bf16(frag16(k0 + 32 + 8 * hi, k0 + 48 + 8 * hi), qf1, st0);
    v8f st1 = wmma_bf16(frag16(k1 + 8 * hi, k1 + 16 + 8 * hi), qf0, vzero);
    st1     = wmma_bf16(frag16(k1 + 32 + 8 * hi, k1 + 48 + 8 * hi), qf1, st1);

    // causal mask + per-lane tile max (lane covers keys {kt + t*16 + r + 8*hi})
    float s0v[8], s1v[8];
    float mt = -1e30f;
#pragma unroll
    for (int r = 0; r < 8; ++r) {
      int key0 = kt + r + 8 * hi;
      float a = (key0 <= q) ? st0[r] * SC : -1e30f;
      float b = (key0 + 16 <= q) ? st1[r] * SC : -1e30f;
      s0v[r] = a; s1v[r] = b;
      mt = fmaxf(mt, fmaxf(a, b));
    }
    mt = fmaxf(mt, __shfl_xor(mt, 16));     // combine the two key half-sets
    float mnew  = fmaxf(mrun, mt);
    float scale = __expf(mrun - mnew);
    float tsum  = 0.f;
    ushort16 up;
#pragma unroll
    for (int r = 0; r < 8; ++r) {
      float p0 = __expf(s0v[r] - mnew);
      float p1 = __expf(s1v[r] - mnew);
      tsum += p0 + p1;
      up[r]     = f2bf(p0);                 // A-frag e<8  <-> S^T tile0 elem r
      up[r + 8] = f2bf(p1);                 // A-frag e>=8 <-> S^T tile1 elem r
    }
    tsum += __shfl_xor(tsum, 16);
    srun = srun * scale + tsum;
    mrun = mnew;

    // rescale O: element r holds query (r + 8*hi); its scale lives in lane r+8*hi
    float osc[8];
#pragma unroll
    for (int r = 0; r < 8; ++r) osc[r] = __shfl(scale, r + 8 * hi);
#pragma unroll
    for (int c = 0; c < 4; ++c) {
#pragma unroll
      for (int r = 0; r < 8; ++r) o[c][r] = o[c][r] * osc[r];
    }

    // O += P * V ; V B-frag: n = dim = c*16+lo, k = key = 8*hi+e / 16+8*hi+e (from vT rows)
    v16bf pf = __builtin_bit_cast(v16bf, up);
#pragma unroll
    for (int c = 0; c < 4; ++c) {
      const unsigned short* vp = vT + (size_t)(c * 16 + lo) * SEQ + kt + 8 * hi;
      o[c] = wmma_bf16(pf, frag16(vp, vp + 16), o[c]);
    }
  }

  float rinv = 1.0f / srun;
  float rnv[8];
#pragma unroll
  for (int r = 0; r < 8; ++r) rnv[r] = __shfl(rinv, r + 8 * hi);
#pragma unroll
  for (int c = 0; c < 4; ++c) {
#pragma unroll
    for (int r = 0; r < 8; ++r) {
      int row = qbase + r + 8 * hi;
      int col = h * HD + c * 16 + lo;
      ctxb[(size_t)row * HID + col] = f2bf(o[c][r] * rnv[r]);
    }
  }
}

// ---------- launch ----------

extern "C" void kernel_launch(void* const* d_in, const int* in_sizes, int n_in,
                              void* d_out, int out_size, void* d_ws, size_t ws_size,
                              hipStream_t stream) {
  const float* hidden  = (const float*)d_in[0];
  const float* w_qkv   = (const float*)d_in[1];
  const float* w_dense = (const float*)d_in[2];
  const int*   pos     = (const int*)d_in[3];
  float* out = (float*)d_out;
  char* ws = (char*)d_ws;

  auto al = [](size_t x) { return (x + 255) & ~(size_t)255; };
  const size_t szW  = (size_t)QKVN * HID * sizeof(unsigned short); // largest wT
  const size_t szHA = (size_t)SEQ * HID * sizeof(unsigned short);
  const size_t szF  = (size_t)SEQ * QKVN * sizeof(float);
  const size_t szK  = (size_t)SEQ * HD * sizeof(unsigned short);

  unsigned short* wT    = (unsigned short*)(ws);
  unsigned short* hA    = (unsigned short*)(ws + al(szW));          // later: qb
  float*          fused = (float*)(ws + al(szW) + al(szHA));        // later: ctxb
  unsigned short* kbuf  = (unsigned short*)(ws + al(szW) + al(szHA) + al(szF));
  unsigned short* vT    = (unsigned short*)(ws + al(szW) + al(szHA) + al(szF) + al(szK));
  unsigned short* qb    = hA;                 // reuse after QKV GEMM consumed hA
  unsigned short* ctxb  = (unsigned short*)fused; // reuse after RoPE consumed fused

  const int nH = SEQ * HID;
  cvt_bf16<<<(nH + 255) / 256, 256, 0, stream>>>(hidden, hA, nH);
  cvt_transpose<<<4096, 256, 0, stream>>>(w_qkv, wT, HID, QKVN);
  gemm_bf16_tn<<<dim3(QKVN / 64, SEQ / 128), 256, 0, stream>>>(hA, wT, fused,
                                                               SEQ, QKVN, HID);
  rope_split<<<SEQ, 256, 0, stream>>>(fused, pos, qb, kbuf, vT);
  cvt_transpose<<<4096, 256, 0, stream>>>(w_dense, wT, HID, HID);
  falcon_attn<<<dim3(SEQ / 128, NHEADS), 256, 0, stream>>>(qb, kbuf, vT, ctxb);
  gemm_bf16_tn<<<dim3(HID / 64, SEQ / 128), 256, 0, stream>>>(ctxb, wT, out,
                                                              SEQ, HID, HID);
}
